// StyleSpaceSythesisBlock_10763188043844
// MI455X (gfx1250) — matrix operand order repaired and checked by reading, not compiled, not run on hardware
//
#include <hip/hip_runtime.h>

typedef __attribute__((ext_vector_type(16))) _Float16 v16h;
typedef __attribute__((ext_vector_type(8)))  _Float16 v8h;
typedef __attribute__((ext_vector_type(8)))  float    v8f;
typedef __attribute__((ext_vector_type(4)))  unsigned int v4u;
typedef __attribute__((ext_vector_type(8)))  int v8i;
typedef __attribute__((ext_vector_type(4)))  int v4i;

#define NB  8
#define C_  512

#if defined(__AMDGCN__) && __has_builtin(__builtin_amdgcn_tensor_load_to_lds) && \
    __has_builtin(__builtin_amdgcn_s_wait_tensorcnt)
#define USE_TDM 1
#else
#define USE_TDM 0
#endif

#if USE_TDM
// Issue one TDM DMA: tile_d1 rows x tile_d0 f16-elements (row stride stride0
// elements) from global 'ga_bytes' into LDS at byte offset 'lds_off'.
// D# per CDNA5 ISA ch.8: group0 {count=1, lds_addr, global_addr, type=2},
// group1 {data_size=2B, tensor dims (huge -> no OOB clip), tile dims, stride}.
__device__ __forceinline__ void tdm_load_2d(unsigned lds_off,
                                            unsigned long long ga_bytes,
                                            int tile_d0, int tile_d1,
                                            long long stride0_elems) {
  v4u g0;
  g0[0] = 1u;                                            // count=1, user D#
  g0[1] = lds_off;                                       // lds_addr [63:32]
  g0[2] = (unsigned)ga_bytes;                            // global_addr lo
  g0[3] = (unsigned)((ga_bytes >> 32) & 0x01FFFFFFu) | 0x80000000u;  // hi | type=2
  v8i g1;
  g1[0] = 0x00010000;                                    // data_size = 2 bytes
  g1[1] = 0;                                             // TD0[15:0] (TD0 = 1<<30)
  g1[2] = 0x4000 | (4 << 16);                            // TD0[31:16], tensor_dim1=4
  g1[3] = (tile_d0 << 16);                               // tile_dim0
  g1[4] = (tile_d1 & 0xFFFF);                            // tile_dim1 (tile_dim2=0)
  g1[5] = (int)(stride0_elems & 0xFFFFFFFFll);           // dim0 stride lo
  g1[6] = (int)((stride0_elems >> 32) & 0xFFFFll);       // dim0 stride hi
  g1[7] = 0;
  v4i gz; gz[0] = gz[1] = gz[2] = gz[3] = 0;             // groups 2/3 unused (2D)
#if __clang_major__ >= 23
  v8i gz8;
#pragma unroll
  for (int i = 0; i < 8; ++i) gz8[i] = 0;
  __builtin_amdgcn_tensor_load_to_lds(g0, g1, gz, gz, gz8, 0);
#else
  __builtin_amdgcn_tensor_load_to_lds(g0, g1, gz, gz, 0);
#endif
}
#endif

// ---------------------------------------------------------------------------
// d[n,o] = rsqrt( sum_i (sum_k w[o,i,k]^2) * s[n,i]^2 + 1e-8 )
// ---------------------------------------------------------------------------
__global__ __launch_bounds__(256) void demod_kernel(const float* __restrict__ w,
                                                    const float* __restrict__ s,
                                                    float* __restrict__ d) {
  int idx = blockIdx.x * 256 + threadIdx.x;
  if (idx >= NB * C_) return;
  int n = idx >> 9, o = idx & 511;
  const float* sn = s + n * C_;
  float acc = 0.f;
  for (int i = 0; i < C_; ++i) {
    float ss = sn[i]; ss *= ss;
    const float* wp = w + ((size_t)o * C_ + i) * 9;
    float wsq = 0.f;
#pragma unroll
    for (int k = 0; k < 9; ++k) wsq += wp[k] * wp[k];
    acc += wsq * ss;
  }
  d[idx] = rsqrtf(acc + 1e-8f);
}

// ---------------------------------------------------------------------------
// xm[n][h][w][c] = f16( x[n][c][h][w] * s0[n][c] )   (NCHW f32 -> NHWC f16)
// ---------------------------------------------------------------------------
__global__ __launch_bounds__(256) void prep_xm_kernel(const float* __restrict__ x,
                                                      const float* __restrict__ s0,
                                                      _Float16* __restrict__ xm) {
  int idx = blockIdx.x * 256 + threadIdx.x;
  int c = idx & 511, w = (idx >> 9) & 31, h = (idx >> 14) & 31, n = idx >> 19;
  float v = x[(((size_t)n * C_ + c) * 32 + h) * 32 + w] * s0[n * C_ + c];
  xm[idx] = (_Float16)v;
}

// ---------------------------------------------------------------------------
// wp1[t][ico][oc][kk] = f16( w1[oc][ico*32+kk][t] )   t = ky*3+kx
// ---------------------------------------------------------------------------
__global__ __launch_bounds__(256) void prep_w1_kernel(const float* __restrict__ w1,
                                                      _Float16* __restrict__ wp1) {
  int idx = blockIdx.x * 256 + threadIdx.x;
  int kk = idx & 31, oc = (idx >> 5) & 511, ico = (idx >> 14) & 15, t = idx >> 18;
  int ic = ico * 32 + kk;
  wp1[idx] = (_Float16)w1[((size_t)oc * C_ + ic) * 9 + t];
}

// ---------------------------------------------------------------------------
// Fused (transposed-conv up2 (3x3) o 4-tap FIR) per-phase 3x3 effective taps:
//   K2[ty,tx] = sum_{fy,fx} 4*fir[fy,fx] * w0[ty+fy-3, tx+fx-3]
//   phase (ry,rx), tap t=e*3+f -> slice K2[ry+4-2e][rx+4-2f]
// ---------------------------------------------------------------------------
__global__ __launch_bounds__(256) void prep_w0_kernel(const float* __restrict__ w0,
                                                      const float* __restrict__ fir,
                                                      _Float16* __restrict__ wp0) {
  int idx = blockIdx.x * 256 + threadIdx.x;
  int kk = idx & 31, oc = (idx >> 5) & 511, ico = (idx >> 14) & 15;
  int rem = idx >> 18;
  int tap = rem % 9, phase = rem / 9;
  int ic = ico * 32 + kk;
  int ry = phase >> 1, rx = phase & 1, e = tap / 3, f = tap % 3;
  int ty = ry + 4 - 2 * e, tx = rx + 4 - 2 * f;
  float acc = 0.f;
#pragma unroll
  for (int fy = 0; fy < 4; ++fy) {
    int wy = ty + fy - 3;
    if (wy < 0 || wy > 2) continue;
#pragma unroll
    for (int fx = 0; fx < 4; ++fx) {
      int wx = tx + fx - 3;
      if (wx < 0 || wx > 2) continue;
      acc += 4.f * fir[fy * 4 + fx] * w0[(((size_t)oc * C_ + ic) * 3 + wy) * 3 + wx];
    }
  }
  wp0[idx] = (_Float16)acc;
}

// ---------------------------------------------------------------------------
// Implicit-GEMM 3x3 conv (pad 1), WMMA f32_16x16x32_f16, K=64 per stage.
//   inp: [B][HW][HW][512] f16 NHWC; wp: [PH][9][16][512][32] f16 packed.
//   Tile: BM spatial x 64 OC per 256-thread / 8-wave32 workgroup.
//   A tile via predicated b128 loads (halo); B tile via TDM DMA to LDS.
// ---------------------------------------------------------------------------
template <int BM, int HW, bool CONV0>
__global__ __launch_bounds__(256) void conv3x3_wmma_kernel(
    const _Float16* __restrict__ inp, const _Float16* __restrict__ wp,
    const float* __restrict__ dmod, const float* __restrict__ noise,
    const float* __restrict__ nscale, const float* __restrict__ bias,
    const float* __restrict__ snext, _Float16* __restrict__ outh,
    float* __restrict__ outf) {
  constexpr int BN = 64;
  constexpr int NT = (BM == 64) ? 2 : 1;       // 16x16 N-tiles per wave
  __shared__ __align__(16) _Float16 lA[2][BM * 64];   // [chunk][m][32]
  __shared__ __align__(16) _Float16 lB[2][BN * 64];   // [chunk][oc'][32]

  const int row = blockIdx.x;        // a (conv0 phase row) or p (conv1 row)
  const int n0 = blockIdx.y * BN;    // oc base
  int bz = blockIdx.z;
  int phase = 0, n;
  if (CONV0) { phase = bz & 3; n = bz >> 2; } else { n = bz; }

  const _Float16* inpN = inp + (size_t)n * HW * HW * C_;
  const _Float16* wpp = wp + (size_t)phase * 9 * 16 * C_ * 32;

  const int tid = threadIdx.x;
  const int lm = tid >> 2;           // 0..63 loader row
  const int lk = (tid & 3) * 8;      // 0,8,16,24 k-part
  const int wv = tid >> 5;           // wave id (wave32)
  const int lane = tid & 31;

  int msub, ng;
  if (BM == 64) { msub = wv & 3; ng = wv >> 2; }
  else          { msub = wv & 1; ng = wv >> 1; }

  v8f acc[NT];
#pragma unroll
  for (int j = 0; j < NT; ++j)
#pragma unroll
    for (int r = 0; r < 8; ++r) acc[j][r] = 0.f;

  const int NSTEP = 9 * 8;           // 9 taps x 8 k-pairs (K=64 per stage)

  auto stage = [&](int s, int buf) {
    int t = s >> 3;                  // tap 0..8
    int icp = s & 7;                 // ic-chunk pair
    int ky = t / 3, kx = t % 3;
    // A tile: BM x 64 channels, zero-filled halo, 2 chunks of 32.
    if (BM == 64 || lm < BM) {
      int m = lm & (BM - 1);
      int yy = row + ky - 1;
      int xx = m + kx - 1;
      bool in = (yy >= 0 && yy < HW && xx >= 0 && xx < HW);
      const _Float16* src = inpN + ((size_t)(yy * HW + xx)) * C_ + icp * 64 + lk;
#pragma unroll
      for (int c = 0; c < 2; ++c) {
        v8h av;
#pragma unroll
        for (int j = 0; j < 8; ++j) av[j] = (_Float16)0.f;
        if (in) av = *(const v8h*)(src + c * 32);
        *(v8h*)&lA[buf][(c * BM + m) * 32 + lk] = av;
      }
      if (in) __builtin_prefetch((const void*)(src + 128), 0, 3);  // k-pair after next
    }
    // B tile: 2 chunks x (64 oc x 32 ch); contiguous 4KB rows, stride 16384 elems
#if USE_TDM
    if (tid == 0) {
      unsigned lds_off = (unsigned)(uintptr_t)(void*)&lB[buf][0];
      unsigned long long ga =
          (unsigned long long)(uintptr_t)(wpp + (((size_t)(t * 16 + icp * 2)) * C_ + n0) * 32);
      tdm_load_2d(lds_off, ga, 2048, 2, (long long)C_ * 32);
    }
#else
#pragma unroll
    for (int c = 0; c < 2; ++c) {
      const _Float16* src =
          wpp + (((size_t)(t * 16 + icp * 2 + c)) * C_ + (n0 + lm)) * 32 + lk;
      *(v8h*)&lB[buf][(c * BN + lm) * 32 + lk] = *(const v8h*)src;
      __builtin_prefetch((const void*)(src + C_ * 32 * 2), 0, 3);
    }
#endif
  };

  stage(0, 0);
  for (int s = 0; s < NSTEP; ++s) {
#if USE_TDM
    __builtin_amdgcn_s_wait_tensorcnt(0);   // B tile DMA for buf[s&1] complete
#endif
    __syncthreads();
    if (s + 1 < NSTEP) stage(s + 1, (s + 1) & 1);
    const _Float16* A = lA[s & 1];
    const _Float16* Bm = lB[s & 1];
    int Ml = msub * 16 + (lane & 15);
    int Kb = (lane >> 4) * 8;        // A: K runs {Kb..Kb+7, Kb+16..Kb+23}
    int KbB = (lane >> 4) * 8;       // B: K runs {KbB..KbB+7, KbB+16..KbB+23}
#pragma unroll
    for (int c = 0; c < 2; ++c) {
      const _Float16* Ac = A + c * BM * 32;
      const _Float16* Bc = Bm + c * BN * 32;
      v16h af;
      *(v8h*)&af = *(const v8h*)&Ac[Ml * 32 + Kb];
      *((v8h*)&af + 1) = *(const v8h*)&Ac[Ml * 32 + 16 + Kb];
#pragma unroll
      for (int j = 0; j < NT; ++j) {
        int Nl = ((NT == 2) ? (ng * 2 + j) : ng) * 16 + (lane & 15);
        v16h bf;
        *(v8h*)&bf = *(const v8h*)&Bc[Nl * 32 + KbB];
        *((v8h*)&bf + 1) = *(const v8h*)&Bc[Nl * 32 + 16 + KbB];
        acc[j] = __builtin_amdgcn_wmma_f32_16x16x32_f16(false, af, false, bf,
                                                        (short)0, acc[j], false, false);
      }
    }
  }

  // Epilogue: demod + noise + bias + lrelu*sqrt2 + clamp (+ s_next, layout)
  const float ns = nscale[0];
#pragma unroll
  for (int j = 0; j < NT; ++j) {
    int nbase = n0 + (((NT == 2) ? (ng * 2 + j) : ng) * 16) + (lane & 15);  // oc
    float dm = dmod[n * C_ + nbase];
    float bs = bias[nbase];
#pragma unroll
    for (int r = 0; r < 8; ++r) {
      int Mloc = msub * 16 + r + 8 * (lane >> 4);
      float v = acc[j][r] * dm;
      if constexpr (!CONV0) {
        int p = row, q = Mloc;
        v += noise[p * 64 + q] * ns + bs;
        v = (v < 0.f ? 0.2f * v : v) * 1.41421356237f;
        v = fminf(fmaxf(v, -256.f), 256.f);
        outf[(((size_t)n * C_ + nbase) * 64 + p) * 64 + q] = v;
      } else {
        int ry = phase >> 1, rx = phase & 1;
        int p = 2 * row + ry, q = 2 * Mloc + rx;
        v += noise[p * 64 + q] * ns + bs;
        v = (v < 0.f ? 0.2f * v : v) * 1.41421356237f;
        v = fminf(fmaxf(v, -256.f), 256.f);
        v *= snext[n * C_ + nbase];   // fold s1 for conv1's A operand
        outh[(((size_t)n * 64 + p) * 64 + q) * C_ + nbase] = (_Float16)v;
      }
    }
  }
}

// ---------------------------------------------------------------------------
// toRGB (1x1, no demod) + FIR-upsampled skip add.
// ---------------------------------------------------------------------------
__global__ __launch_bounds__(256) void torgb_skip_kernel(
    const float* __restrict__ z, const float* __restrict__ img,
    const float* __restrict__ s2, const float* __restrict__ wt,
    const float* __restrict__ bt, const float* __restrict__ fir,
    float* __restrict__ outimg) {
  int idx = blockIdx.x * 256 + threadIdx.x;  // 8*64*64
  int q = idx & 63, p = (idx >> 6) & 63, n = idx >> 12;
  const float inv = 0.04419417382415922f;    // 1/sqrt(512)
  float a0 = 0.f, a1 = 0.f, a2 = 0.f;
  const float* zp = z + (((size_t)n * C_) * 64 + p) * 64 + q;
  const float* s2n = s2 + n * C_;
  for (int i = 0; i < C_; ++i) {
    float v = zp[(size_t)i * 4096] * (s2n[i] * inv);
    a0 += v * wt[i];
    a1 += v * wt[C_ + i];
    a2 += v * wt[2 * C_ + i];
  }
  float rgb[3] = {a0 + bt[0], a1 + bt[1], a2 + bt[2]};
#pragma unroll
  for (int c = 0; c < 3; ++c) rgb[c] = fminf(fmaxf(rgb[c], -256.f), 256.f);
#pragma unroll
  for (int c = 0; c < 3; ++c) {
    float sk = 0.f;
#pragma unroll
    for (int fy = 0; fy < 4; ++fy) {
      int u = p + fy - 2;
      if (u < 0 || u >= 64 || (u & 1)) continue;
      int h = u >> 1;
#pragma unroll
      for (int fx = 0; fx < 4; ++fx) {
        int vv = q + fx - 2;
        if (vv < 0 || vv >= 64 || (vv & 1)) continue;
        int w = vv >> 1;
        sk += 4.f * fir[fy * 4 + fx] * img[(((size_t)n * 3 + c) * 32 + h) * 32 + w];
      }
    }
    outimg[(((size_t)n * 3 + c) * 64 + p) * 64 + q] = sk + rgb[c];
  }
}

// ---------------------------------------------------------------------------
extern "C" void kernel_launch(void* const* d_in, const int* in_sizes, int n_in,
                              void* d_out, int out_size, void* d_ws, size_t ws_size,
                              hipStream_t stream) {
  (void)in_sizes; (void)n_in; (void)out_size; (void)ws_size;
  const float* x   = (const float*)d_in[0];
  const float* img = (const float*)d_in[1];
  const float* s0  = (const float*)d_in[2];
  const float* s1  = (const float*)d_in[3];
  const float* s2  = (const float*)d_in[4];
  const float* w0  = (const float*)d_in[5];
  const float* b0  = (const float*)d_in[6];
  const float* nc0 = (const float*)d_in[7];
  const float* ns0 = (const float*)d_in[8];
  const float* w1  = (const float*)d_in[9];
  const float* b1  = (const float*)d_in[10];
  const float* nc1 = (const float*)d_in[11];
  const float* ns1 = (const float*)d_in[12];
  const float* wt  = (const float*)d_in[13];
  const float* bt  = (const float*)d_in[14];
  const float* fir = (const float*)d_in[15];

  char* ws = (char*)d_ws;
  float*    d0  = (float*)(ws + 0);               // 16 KB
  float*    d1  = (float*)(ws + 16384);           // 16 KB
  _Float16* xm  = (_Float16*)(ws + 32768);        // 8.39 MB  [8][32][32][512]
  _Float16* wp0 = (_Float16*)(ws + 8421376);      // 18.87 MB [4][9][16][512][32]
  _Float16* wp1 = (_Float16*)(ws + 27295744);     // 4.72 MB  [9][16][512][32]
  _Float16* ym  = (_Float16*)(ws + 32014336);     // 33.55 MB [8][64][64][512]

  float* zout = (float*)d_out;                    // [8][512][64][64]
  float* iout = zout + (size_t)NB * C_ * 64 * 64; // [8][3][64][64]

  demod_kernel<<<16, 256, 0, stream>>>(w0, s0, d0);
  demod_kernel<<<16, 256, 0, stream>>>(w1, s1, d1);
  prep_xm_kernel<<<16384, 256, 0, stream>>>(x, s0, xm);
  prep_w0_kernel<<<36864, 256, 0, stream>>>(w0, fir, wp0);
  prep_w1_kernel<<<9216, 256, 0, stream>>>(w1, wp1);

  conv3x3_wmma_kernel<32, 32, true><<<dim3(32, 8, 32), 256, 0, stream>>>(
      xm, wp0, d0, nc0, ns0, b0, s1, ym, nullptr);
  conv3x3_wmma_kernel<64, 64, false><<<dim3(64, 8, 8), 256, 0, stream>>>(
      ym, wp1, d1, nc1, ns1, b1, nullptr, nullptr, zout);

  torgb_skip_kernel<<<128, 256, 0, stream>>>(zout, img, s2, wt, bt, fir, iout);
}
